// MultiHeadGAT_23347442221667
// MI455X (gfx1250) — matrix operand order repaired and checked by reading, not compiled, not run on hardware
//
#include <hip/hip_runtime.h>
#include <hip/hip_bf16.h>

typedef __attribute__((ext_vector_type(16))) _Float16 v16h;
typedef __attribute__((ext_vector_type(8)))  _Float16 v8h;
typedef __attribute__((ext_vector_type(8)))  float    v8f;
typedef __attribute__((ext_vector_type(4)))  float    v4f;
typedef __attribute__((ext_vector_type(4)))  int      v4i;

#define WMMA_F32_F16(a,b,c) \
  __builtin_amdgcn_wmma_f32_16x16x32_f16(false,(a),false,(b),(short)0,(c),false,false)

// Problem constants: B=8, N=1024, D=128, H=8, HD=16, rows = B*N = 8192

// ---------------------------------------------------------------------------
// Prep: convert FFN weights to f16 once (tiny, L2-resident afterwards)
// ---------------------------------------------------------------------------
__global__ void k_w2h(const float* __restrict__ w1, const float* __restrict__ w2,
                      _Float16* __restrict__ w1h, _Float16* __restrict__ w2h) {
  int i = blockIdx.x * blockDim.x + threadIdx.x;   // 32768 threads
  if (i < 128 * 256) w1h[i] = (_Float16)w1[i];
  if (i < 256 * 128) w2h[i] = (_Float16)w2[i];
}

// ---------------------------------------------------------------------------
// Kernel 1: Wh = h @ W  (8192 x 128 x 128), one wave per 16x16 tile of head hh
// Output stored as f16 [8192][128] (head-major columns h*16+d) for reuse as
// the WMMA B-operand in the attention aggregation.
// ---------------------------------------------------------------------------
__global__ void __launch_bounds__(32) k_proj(const float* __restrict__ hin,
                                             const float* __restrict__ W,
                                             _Float16* __restrict__ whf16) {
  const int lane = threadIdx.x & 31;
  const int l16 = lane & 15, hi = lane >> 4;
  const int mtile = blockIdx.x;   // 0..511
  const int hh    = blockIdx.y;   // 0..7
  const float* hrow = hin + (size_t)(mtile * 16 + l16) * 128;  // A row = l16
  v8f c = {};
#pragma unroll
  for (int ko = 0; ko < 128; ko += 32) {
    // A-operand (16x32 f16): lane l16 holds K = ko+8*hi+{0..7} and ko+16+8*hi+{0..7}
    v4f a0 = *(const v4f*)(hrow + ko + 8 * hi);
    v4f a1 = *(const v4f*)(hrow + ko + 8 * hi + 4);
    v4f a2 = *(const v4f*)(hrow + ko + 16 + 8 * hi);
    v4f a3 = *(const v4f*)(hrow + ko + 16 + 8 * hi + 4);
    v16h A;
#pragma unroll
    for (int j = 0; j < 4; ++j) {
      A[j] = (_Float16)a0[j];  A[4 + j]  = (_Float16)a1[j];
      A[8 + j] = (_Float16)a2[j]; A[12 + j] = (_Float16)a3[j];
    }
    // B-operand (32x16 f16): lane = column l16, 16 halves = K in table order
    const float* wrow = W + ((size_t)hh * 128 + ko) * 16 + l16;
    v16h Bv;
#pragma unroll
    for (int j = 0; j < 16; ++j) Bv[j] = (_Float16)wrow[(size_t)((j & 7) + 8 * hi + 16 * (j >> 3)) * 16];
    c = WMMA_F32_F16(A, Bv, c);
  }
#pragma unroll
  for (int i = 0; i < 8; ++i) {   // C: row = i+8*hi, col = l16
    int m = mtile * 16 + i + 8 * hi;
    whf16[(size_t)m * 128 + hh * 16 + l16] = (_Float16)c[i];
  }
}

// ---------------------------------------------------------------------------
// Kernel 2: s_src[b,h,n] = Wh . a_src ; s_dst = Wh . a_dst  (65536 threads)
// ---------------------------------------------------------------------------
__global__ void k_scores(const _Float16* __restrict__ whf16,
                         const float* __restrict__ a,
                         float* __restrict__ ssrc, float* __restrict__ sdst) {
  int idx = blockIdx.x * blockDim.x + threadIdx.x;   // (b*8+h)*1024 + n
  int n = idx & 1023, bh = idx >> 10;
  int b = bh >> 3, hh = bh & 7;
  const _Float16* w = whf16 + (size_t)(b * 1024 + n) * 128 + hh * 16;
  const float* av = a + hh * 32;
  float s0 = 0.f, s1 = 0.f;
#pragma unroll
  for (int d = 0; d < 16; ++d) {
    float x = (float)w[d];
    s0 += x * av[d];
    s1 += x * av[16 + d];
  }
  ssrc[idx] = s0; sdst[idx] = s1;
}

// ---------------------------------------------------------------------------
// Kernel 3: fused masked-leakyReLU-softmax + (alpha @ Wh) via WMMA.
// One wave per (b, h, 16-row tile); e never materialized. BOTH passes run in
// the WMMA A-operand layout: lane owns row l16, K-positions 8*hi+{0..7} and
// 16+8*hi+{0..7} of each 32-wide column chunk.
//   Pass 1: load adj runs (4x b128) + s_j runs, compute masked leaky e,
//           pack 16 ok-bits/lane/chunk into LDS (4KB), track per-lane max.
//           Row max = single shfl_xor(16) (lane pair covers whole row).
//   Pass 2: rebuild p = exp(e - rowmax) masked, accumulate unnormalized
//           p @ Wh with one WMMA per chunk; normalize by rowsum at the end.
// ---------------------------------------------------------------------------
__global__ void __launch_bounds__(32) k_attn(const int* __restrict__ adj,
                                             const _Float16* __restrict__ whf16,
                                             const float* __restrict__ ssrc,
                                             const float* __restrict__ sdst,
                                             float* __restrict__ hcat) {
  __shared__ unsigned okb[32 * 32];   // [chunk][lane] packed ok bits
  const int lane = threadIdx.x & 31;
  const int l16 = lane & 15, hi = lane >> 4;
  const int tile = blockIdx.x, hh = blockIdx.y, b = blockIdx.z;
  const int row0 = tile * 16;
  const int bh = b * 8 + hh;
  const float* sj_base = sdst + (size_t)bh * 1024;
  const float NEGI = -3.0e38f;
  float si_me = ssrc[(size_t)bh * 1024 + row0 + l16];
  const int* adjrow = adj + ((size_t)(b * 1024 + row0 + l16)) * 1024;

  // ---- pass 1: masked row max + ok-bit packing ----
  float pmax = NEGI;
  for (int ch = 0; ch < 32; ++ch) {
    const int mb = ch * 32;
    if (ch < 31)  // pull next chunk's adj line toward the caches
      __builtin_prefetch((const void*)(adjrow + mb + 32 + 8 * hi), 0, 1);
    v4i a0 = *(const v4i*)(adjrow + mb + 8 * hi);
    v4i a1 = *(const v4i*)(adjrow + mb + 8 * hi + 4);
    v4i a2 = *(const v4i*)(adjrow + mb + 16 + 8 * hi);
    v4i a3 = *(const v4i*)(adjrow + mb + 16 + 8 * hi + 4);
    v4f j0 = *(const v4f*)(sj_base + mb + 8 * hi);
    v4f j1 = *(const v4f*)(sj_base + mb + 8 * hi + 4);
    v4f j2 = *(const v4f*)(sj_base + mb + 16 + 8 * hi);
    v4f j3 = *(const v4f*)(sj_base + mb + 16 + 8 * hi + 4);
    unsigned bits = 0;
#pragma unroll
    for (int j = 0; j < 8; ++j) {
      int av  = (j < 4) ? a0[j & 3] : a1[j & 3];
      float sj = (j < 4) ? j0[j & 3] : j1[j & 3];
      float e = si_me + sj; e = (e >= 0.f) ? e : 0.2f * e;
      bool ok = av != 0;
      bits |= (unsigned)ok << j;
      pmax = fmaxf(pmax, ok ? e : NEGI);
    }
#pragma unroll
    for (int j = 0; j < 8; ++j) {
      int av  = (j < 4) ? a2[j & 3] : a3[j & 3];
      float sj = (j < 4) ? j2[j & 3] : j3[j & 3];
      float e = si_me + sj; e = (e >= 0.f) ? e : 0.2f * e;
      bool ok = av != 0;
      bits |= (unsigned)ok << (8 + j);
      pmax = fmaxf(pmax, ok ? e : NEGI);
    }
    okb[ch * 32 + lane] = bits;
  }
  float rm = fmaxf(pmax, __shfl_xor(pmax, 16, 32));  // rowmax for row l16
  __syncthreads();

  // ---- pass 2: unnormalized softmax @ Wh via WMMA ----
  v8f cacc = {};
  float psum = 0.f;
  for (int ch = 0; ch < 32; ++ch) {
    const int mb = ch * 32;
    unsigned bits = okb[ch * 32 + lane];
    v4f j0 = *(const v4f*)(sj_base + mb + 8 * hi);
    v4f j1 = *(const v4f*)(sj_base + mb + 8 * hi + 4);
    v4f j2 = *(const v4f*)(sj_base + mb + 16 + 8 * hi);
    v4f j3 = *(const v4f*)(sj_base + mb + 16 + 8 * hi + 4);
    v16h A;
#pragma unroll
    for (int j = 0; j < 8; ++j) {
      float sj = (j < 4) ? j0[j & 3] : j1[j & 3];
      float e = si_me + sj; e = (e >= 0.f) ? e : 0.2f * e;
      float p = ((bits >> j) & 1u) ? __expf(e - rm) : 0.f;
      psum += p; A[j] = (_Float16)p;
    }
#pragma unroll
    for (int j = 0; j < 8; ++j) {
      float sj = (j < 4) ? j2[j & 3] : j3[j & 3];
      float e = si_me + sj; e = (e >= 0.f) ? e : 0.2f * e;
      float p = ((bits >> (8 + j)) & 1u) ? __expf(e - rm) : 0.f;
      psum += p; A[8 + j] = (_Float16)p;
    }
    const _Float16* Bc = whf16 + ((size_t)(b * 1024 + mb)) * 128 + hh * 16 + l16;
    v16h Bv;
#pragma unroll
    for (int j = 0; j < 16; ++j) Bv[j] = Bc[(size_t)((j & 7) + 8 * hi + 16 * (j >> 3)) * 128];
    cacc = WMMA_F32_F16(A, Bv, cacc);
  }
  // rowsum for row l16 = this lane's partial + its pair lane (xor 16)
  float rs = psum + __shfl_xor(psum, 16, 32);
#pragma unroll
  for (int i = 0; i < 8; ++i) {
    int q = i + 8 * hi;                 // C row
    float s = __shfl(rs, q, 32);        // lane q holds rowsum[q]
    float v = (s > 0.f) ? (cacc[i] / s) : 0.f;   // row_valid==false -> 0
    hcat[((size_t)(b * 1024 + row0 + q)) * 128 + hh * 16 + l16] = v;
  }
}

// ---------------------------------------------------------------------------
// Kernel 4: residual + LayerNorm1 -> f32 (for second residual) + f16 (GEMM A)
// One wave per row (8 rows / 256-thread block).
// ---------------------------------------------------------------------------
__global__ void k_ln1(const float* __restrict__ att, const float* __restrict__ hin,
                      const float* __restrict__ g, const float* __restrict__ bb,
                      float* __restrict__ of32, _Float16* __restrict__ of16) {
  int row = blockIdx.x * (blockDim.x >> 5) + (threadIdx.x >> 5);
  int lane = threadIdx.x & 31;
  const size_t base = (size_t)row * 128;
  float x[4]; float s = 0.f;
#pragma unroll
  for (int j = 0; j < 4; ++j) {
    int col = j * 32 + lane;
    x[j] = att[base + col] + hin[base + col];
    s += x[j];
  }
#pragma unroll
  for (int off = 16; off >= 1; off >>= 1) s += __shfl_xor(s, off, 32);
  float mu = s * (1.f / 128.f);
  float v = 0.f;
#pragma unroll
  for (int j = 0; j < 4; ++j) { float d = x[j] - mu; v += d * d; }
#pragma unroll
  for (int off = 16; off >= 1; off >>= 1) v += __shfl_xor(v, off, 32);
  float rstd = rsqrtf(v * (1.f / 128.f) + 1e-5f);
#pragma unroll
  for (int j = 0; j < 4; ++j) {
    int col = j * 32 + lane;
    float y = (x[j] - mu) * rstd * g[col] + bb[col];
    of32[base + col] = y;
    of16[base + col] = (_Float16)y;
  }
}

// ---------------------------------------------------------------------------
// Kernel 5: FFN GEMM1 (8192x128x256) + bias + ReLU -> f16, wave per 16x16 tile
// ---------------------------------------------------------------------------
__global__ void __launch_bounds__(32) k_ffn1(const _Float16* __restrict__ xa,
                                             const _Float16* __restrict__ w1h,
                                             const float* __restrict__ b1,
                                             _Float16* __restrict__ t1) {
  const int lane = threadIdx.x & 31, l16 = lane & 15, hi = lane >> 4;
  const int mtile = blockIdx.x, ntile = blockIdx.y;
  const _Float16* arow = xa + (size_t)(mtile * 16 + l16) * 128;
  v8f c = {};
#pragma unroll
  for (int ko = 0; ko < 128; ko += 32) {
    v8h r0 = *(const v8h*)(arow + ko + 8 * hi);
    v8h r1 = *(const v8h*)(arow + ko + 16 + 8 * hi);
    v16h A;
#pragma unroll
    for (int j = 0; j < 8; ++j) { A[j] = r0[j]; A[8 + j] = r1[j]; }
    const _Float16* Bc = w1h + (size_t)ko * 256 + ntile * 16 + l16;
    v16h Bv;
#pragma unroll
    for (int j = 0; j < 16; ++j) Bv[j] = Bc[(size_t)((j & 7) + 8 * hi + 16 * (j >> 3)) * 256];
    c = WMMA_F32_F16(A, Bv, c);
  }
  float bias = b1[ntile * 16 + l16];
#pragma unroll
  for (int i = 0; i < 8; ++i) {
    float v = c[i] + bias;
    v = v > 0.f ? v : 0.f;
    t1[(size_t)(mtile * 16 + i + 8 * hi) * 256 + ntile * 16 + l16] = (_Float16)v;
  }
}

// ---------------------------------------------------------------------------
// Kernel 6: FFN GEMM2 (8192x256x128) + bias + residual + LayerNorm2, fused.
// One wave computes a full 16x128 strip (8 accumulators) so LN2 row
// reductions stay in-wave (shfl within 16-lane halves).
// ---------------------------------------------------------------------------
__global__ void __launch_bounds__(32) k_ffn2_ln2(const _Float16* __restrict__ t1,
                                                 const _Float16* __restrict__ w2h,
                                                 const float* __restrict__ b2,
                                                 const float* __restrict__ xres,
                                                 const float* __restrict__ g2,
                                                 const float* __restrict__ bb2,
                                                 float* __restrict__ out) {
  const int lane = threadIdx.x & 31, l16 = lane & 15, hi = lane >> 4;
  const int mtile = blockIdx.x;
  const _Float16* arow = t1 + (size_t)(mtile * 16 + l16) * 256;
  v8f acc[8] = {};
#pragma unroll
  for (int ko = 0; ko < 256; ko += 32) {
    v8h r0 = *(const v8h*)(arow + ko + 8 * hi);
    v8h r1 = *(const v8h*)(arow + ko + 16 + 8 * hi);
    v16h A;
#pragma unroll
    for (int j = 0; j < 8; ++j) { A[j] = r0[j]; A[8 + j] = r1[j]; }
#pragma unroll
    for (int t = 0; t < 8; ++t) {
      const _Float16* Bc = w2h + (size_t)ko * 128 + t * 16 + l16;
      v16h Bv;
#pragma unroll
      for (int j = 0; j < 16; ++j) Bv[j] = Bc[(size_t)((j & 7) + 8 * hi + 16 * (j >> 3)) * 128];
      acc[t] = WMMA_F32_F16(A, Bv, acc[t]);
    }
  }
  float x[8][8];   // [ntile][row i]
#pragma unroll
  for (int t = 0; t < 8; ++t) {
    float bias = b2[t * 16 + l16];
#pragma unroll
    for (int i = 0; i < 8; ++i) {
      int m = mtile * 16 + i + 8 * hi;
      x[t][i] = acc[t][i] + bias + xres[(size_t)m * 128 + t * 16 + l16];
    }
  }
  float mu[8], rstd[8];
#pragma unroll
  for (int i = 0; i < 8; ++i) {
    float s = 0.f;
#pragma unroll
    for (int t = 0; t < 8; ++t) s += x[t][i];
#pragma unroll
    for (int off = 8; off >= 1; off >>= 1) s += __shfl_xor(s, off, 32);
    mu[i] = s * (1.f / 128.f);
    float v = 0.f;
#pragma unroll
    for (int t = 0; t < 8; ++t) { float d = x[t][i] - mu[i]; v += d * d; }
#pragma unroll
    for (int off = 8; off >= 1; off >>= 1) v += __shfl_xor(v, off, 32);
    rstd[i] = rsqrtf(v * (1.f / 128.f) + 1e-5f);
  }
#pragma unroll
  for (int t = 0; t < 8; ++t) {
    float gv = g2[t * 16 + l16], bv = bb2[t * 16 + l16];
#pragma unroll
    for (int i = 0; i < 8; ++i) {
      int m = mtile * 16 + i + 8 * hi;
      out[(size_t)m * 128 + t * 16 + l16] = (x[t][i] - mu[i]) * rstd[i] * gv + bv;
    }
  }
}

// ---------------------------------------------------------------------------
extern "C" void kernel_launch(void* const* d_in, const int* in_sizes, int n_in,
                              void* d_out, int out_size, void* d_ws, size_t ws_size,
                              hipStream_t stream) {
  const float* hin = (const float*)d_in[0];
  const int*   adj = (const int*)d_in[1];
  const float* W   = (const float*)d_in[2];
  const float* a   = (const float*)d_in[3];
  const float* g1  = (const float*)d_in[4];
  const float* bb1 = (const float*)d_in[5];
  const float* w1  = (const float*)d_in[6];
  const float* b1  = (const float*)d_in[7];
  const float* w2  = (const float*)d_in[8];
  const float* b2  = (const float*)d_in[9];
  const float* g2  = (const float*)d_in[10];
  const float* bb2 = (const float*)d_in[11];
  float* out = (float*)d_out;

  char* ws = (char*)d_ws;
  size_t off = 0;
  auto alloc = [&](size_t bytes) -> void* {
    void* p = ws + off;
    off += (bytes + 255) & ~(size_t)255;
    return p;
  };
  _Float16* whf16 = (_Float16*)alloc((size_t)8192 * 128 * 2);
  float*    ssrc  = (float*)   alloc((size_t)65536 * 4);
  float*    sdst  = (float*)   alloc((size_t)65536 * 4);
  float*    att   = (float*)   alloc((size_t)8192 * 128 * 4);
  float*    x32   = (float*)   alloc((size_t)8192 * 128 * 4);
  _Float16* x16   = (_Float16*)alloc((size_t)8192 * 128 * 2);
  _Float16* t1    = (_Float16*)alloc((size_t)8192 * 256 * 2);
  _Float16* w1h   = (_Float16*)alloc((size_t)128 * 256 * 2);
  _Float16* w2h   = (_Float16*)alloc((size_t)256 * 128 * 2);

  k_w2h<<<dim3(128), dim3(256), 0, stream>>>(w1, w2, w1h, w2h);
  k_proj<<<dim3(512, 8), dim3(32), 0, stream>>>(hin, W, whf16);
  k_scores<<<dim3(256), dim3(256), 0, stream>>>(whf16, a, ssrc, sdst);
  k_attn<<<dim3(64, 8, 8), dim3(32), 0, stream>>>(adj, whf16, ssrc, sdst, att);
  k_ln1<<<dim3(1024), dim3(256), 0, stream>>>(att, hin, g1, bb1, x32, x16);
  k_ffn1<<<dim3(512, 16), dim3(32), 0, stream>>>(x16, w1h, b1, t1);
  k_ffn2_ln2<<<dim3(512), dim3(32), 0, stream>>>(t1, w2h, b2, x32, g2, bb2, out);
}
